// GroundingBlock_18751827214457
// MI455X (gfx1250) — matrix operand order repaired and checked by reading, not compiled, not run on hardware
//
#include <hip/hip_runtime.h>
#include <cstdint>
#include <cstddef>

// Problem constants (from reference)
#define B_    4
#define CH_   256
#define CL_   128
#define N_    4096     // H*W = 64*64
#define FFN_  256
#define EPS_  1e-5f

typedef __bf16 bf16_t;
typedef bf16_t bf16x8  __attribute__((ext_vector_type(8)));
typedef bf16_t bf16x16 __attribute__((ext_vector_type(16)));
typedef float  f32x8   __attribute__((ext_vector_type(8)));

// Build a 16-element bf16 WMMA fragment from two 16B (8-elem) chunks.
__device__ __forceinline__ bf16x16 ld_frag16(const bf16_t* lo, const bf16_t* hi) {
  bf16x8 a = *(const bf16x8*)lo;
  bf16x8 b = *(const bf16x8*)hi;
  bf16x16 r;
#pragma unroll
  for (int i = 0; i < 8; ++i) { r[i] = a[i]; r[i + 8] = b[i]; }
  return r;
}

// ---- DPP16 rotation-based 16-lane reductions (single v_mov_b32_dpp/step) --
// Combining row_ror:1,2,4,8 gives every lane the reduction over its full
// 16-lane row (windows 1->2->4->8->16 contiguous lanes mod 16). Both wave32
// halves reduce independently, matching the WMMA C-layout rows.
template <int ROR>
__device__ __forceinline__ float dpp_ror(float v) {
  constexpr int ctrl = 0x120 + ROR;  // DPP16 ROW_ROR:n
  return __int_as_float(__builtin_amdgcn_update_dpp(0, __float_as_int(v),
                                                    ctrl, 0xF, 0xF, true));
}
__device__ __forceinline__ float rowred_max(float v) {
  v = fmaxf(v, dpp_ror<1>(v));
  v = fmaxf(v, dpp_ror<2>(v));
  v = fmaxf(v, dpp_ror<4>(v));
  v = fmaxf(v, dpp_ror<8>(v));
  return v;
}
__device__ __forceinline__ float rowred_sum(float v) {
  v += dpp_ror<1>(v);
  v += dpp_ror<2>(v);
  v += dpp_ror<4>(v);
  v += dpp_ror<8>(v);
  return v;
}

// ---------------- fp32 -> bf16 convert (weights, row-major kept) -----------
__global__ void cvt_bf16_kernel(const float* __restrict__ in, bf16_t* __restrict__ out, int n) {
  int i = blockIdx.x * blockDim.x + threadIdx.x;
  if (i < n) out[i] = (bf16_t)in[i];
}

// ---------------- GroupNorm(groups=1) stats: per-sample mean & rstd --------
__global__ __launch_bounds__(256) void stats_kernel(const float* __restrict__ x,
                                                    int perSample, float* __restrict__ out2) {
  __shared__ float s1[256], s2[256];
  const int b = blockIdx.x;
  x += (size_t)b * perSample;
  float s = 0.f, ss = 0.f;
  for (int i = threadIdx.x; i < perSample; i += 256) {
    float v = x[i];
    s += v; ss += v * v;
  }
  s1[threadIdx.x] = s; s2[threadIdx.x] = ss;
  __syncthreads();
  for (int st = 128; st > 0; st >>= 1) {
    if (threadIdx.x < st) {
      s1[threadIdx.x] += s1[threadIdx.x + st];
      s2[threadIdx.x] += s2[threadIdx.x + st];
    }
    __syncthreads();
  }
  if (threadIdx.x == 0) {
    float inv = 1.f / (float)perSample;
    float mu  = s1[0] * inv;
    float var = s2[0] * inv - mu * mu;
    out2[b * 2 + 0] = mu;
    out2[b * 2 + 1] = rsqrtf(var + EPS_);
  }
}

// ------- normalize + affine, write channels-last bf16: out[b][n][c] --------
// i enumerates the OUTPUT order (c fastest) so stores are fully coalesced;
// the strided fp32 reads hit L2 (everything here is L2-resident).
__global__ void norm_t_kernel(const float* __restrict__ x /*[b][c][n]*/,
                              const float* __restrict__ st,
                              const float* __restrict__ w, const float* __restrict__ bias,
                              bf16_t* __restrict__ out /*[b][n][c]*/, int C, int total) {
  int i = blockIdx.x * blockDim.x + threadIdx.x;
  if (i >= total) return;
  int b = i / (N_ * C);
  int rem = i - b * (N_ * C);
  int n = rem / C;
  int c = rem - n * C;
  float mu = st[b * 2], rs = st[b * 2 + 1];
  float v = x[((size_t)b * C + c) * N_ + n];
  out[i] = (bf16_t)(((v - mu) * rs) * w[c] + bias[c]);
}

// ---------------- LDS-free bf16 GEMM:  C[M,N] = W[M,K] @ X[K,N] ------------
// X is stored TRANSPOSED (channels-last): Xt[b][n][K], so the B-fragment
// (lane = column n, 16 K-contiguous values) is two direct 16B global loads.
// A-fragment comes straight from the row-major weight. No LDS, no barriers.
// Block: 256 thr = 8 waves as 4(m) x 2(n); wave tile 16M x 64N.
// EPI: 0 = bf16 store [m][n] (for V, which attention wants d-major)
//      1 = bf16*scale store transposed [n][m] (packed 16B stores)
//      2 = exact GELU, bf16 transposed [n][m]
//      3 = fp32: out = resid + gamma*acc, layout [m][n]
template <int EPI>
__global__ __launch_bounds__(256) void gemm_tn_kernel(
    const bf16_t* __restrict__ Wm, const bf16_t* __restrict__ Xt,
    int M, int K, int N,
    bf16_t* __restrict__ outb, float* outf,
    const float* resid, const float* __restrict__ gamma, float scalep) {
  const int lane = threadIdx.x & 31, wid = threadIdx.x >> 5;
  const int half = lane >> 4, l15 = lane & 15;
  const int wm = wid >> 1, wn = wid & 1;
  const int m0 = blockIdx.y * 64 + wm * 16;
  const int n0 = blockIdx.x * 128 + wn * 64;
  const int bz = blockIdx.z;
  Xt += (size_t)bz * N * K;

  f32x8 acc[4];
#pragma unroll
  for (int j = 0; j < 4; ++j)
#pragma unroll
    for (int r = 0; r < 8; ++r) acc[j][r] = 0.f;

  const bf16_t* arow = Wm + (size_t)(m0 + l15) * K + half * 16;
  const bf16_t* brow[4];
#pragma unroll
  for (int j = 0; j < 4; ++j)
    brow[j] = Xt + (size_t)(n0 + j * 16 + l15) * K + half * 16;

  for (int k0 = 0; k0 < K; k0 += 32) {
    bf16x16 afrag = ld_frag16(arow + k0, arow + k0 + 8);
#pragma unroll
    for (int j = 0; j < 4; ++j) {
      bf16x16 bfrag = ld_frag16(brow[j] + k0, brow[j] + k0 + 8);
      acc[j] = __builtin_amdgcn_wmma_f32_16x16x32_bf16(false, afrag, false, bfrag,
                                                       (short)0, acc[j], false, false);
    }
  }

  if constexpr (EPI == 0) {
    const size_t obase = (size_t)bz * M * N;
#pragma unroll
    for (int j = 0; j < 4; ++j) {
      int col = n0 + j * 16 + l15;
#pragma unroll
      for (int r = 0; r < 8; ++r) {
        int row = m0 + r + half * 8;
        outb[obase + (size_t)row * N + col] = (bf16_t)acc[j][r];
      }
    }
  } else if constexpr (EPI == 1 || EPI == 2) {
    // transposed, packed: lane's 8 rows are M-contiguous -> one 16B store
#pragma unroll
    for (int j = 0; j < 4; ++j) {
      int col = n0 + j * 16 + l15;
      bf16x8 pk;
#pragma unroll
      for (int r = 0; r < 8; ++r) {
        float vv = acc[j][r];
        if constexpr (EPI == 2)
          vv = 0.5f * vv * (1.f + erff(vv * 0.70710678118f));  // exact GELU
        else
          vv *= scalep;
        pk[r] = (bf16_t)vv;
      }
      *(bf16x8*)(outb + ((size_t)bz * N + col) * M + m0 + half * 8) = pk;
    }
  } else {  // EPI == 3
    const size_t obase = (size_t)bz * M * N;
    const float g = gamma[0];
#pragma unroll
    for (int j = 0; j < 4; ++j) {
      int col = n0 + j * 16 + l15;
#pragma unroll
      for (int r = 0; r < 8; ++r) {
        int row = m0 + r + half * 8;
        size_t idx = obase + (size_t)row * N + col;
        outf[idx] = resid[idx] + g * acc[j][r];
      }
    }
  }
}

// ---------------- Flash attention, bf16 WMMA, d=32, 64-col KV tiles --------
// qt/kt: [b][n][128] channels-last (Q pre-scaled by 1/sqrt(d) in its GEMM).
// v:     [b][c][n]  d-major, so the O-update B-fragment (K-dim = n) is
//        memory-contiguous -> direct global loads.
// ot:    [b][n][128] channels-last (feeds the proj GEMM directly).
// 8 waves x 16 q-rows = 128 rows/block; no barriers: the only LDS is the
// wave-private P scratch used for the C-layout -> A-layout reshuffle.
__global__ __launch_bounds__(256) void attn_kernel(
    const bf16_t* __restrict__ qt, const bf16_t* __restrict__ kt,
    const bf16_t* __restrict__ v, bf16_t* __restrict__ ot) {
  __shared__ __align__(16) bf16_t plds_all[8 * 16 * 80];  // [wave][m=16][k=64], pitch 80 halves
  const int lane = threadIdx.x & 31, wid = threadIdx.x >> 5;
  const int half = lane >> 4, l15 = lane & 15;
  const int bh = blockIdx.y;          // b*4 + h
  const int b = bh >> 2, h = bh & 3;
  const int nq0 = blockIdx.x * 128;
  bf16_t* plds = plds_all + wid * (16 * 80);

  // Q A-fragment: direct global (row nq, 32 d-contiguous), lives in regs
  const bf16_t* qrow = qt + ((size_t)b * N_ + nq0 + wid * 16 + l15) * CL_ + h * 32 + half * 16;
  bf16x16 aq = ld_frag16(qrow, qrow + 8);

  const bf16_t* khead = kt + (size_t)b * N_ * CL_ + h * 32;
  const bf16_t* vhead = v + ((size_t)b * CL_ + h * 32) * N_;

  f32x8 o0, o1;
  float mo[8], lo[8];
#pragma unroll
  for (int r = 0; r < 8; ++r) { o0[r] = 0.f; o1[r] = 0.f; mo[r] = -3.0e38f; lo[r] = 0.f; }

  for (int n0 = 0; n0 < N_; n0 += 64) {
    // prefetch next KV tile into caches (global_prefetch_b8)
    if (n0 + 64 < N_) {
      __builtin_prefetch(khead + (size_t)(n0 + 64 + lane) * CL_, 0, 0);
      __builtin_prefetch(vhead + (size_t)lane * N_ + n0 + 64, 0, 0);
    }

    // S = Q*K^T over d=32: 4 WMMAs (16x64 tile), K B-fragments direct global
    f32x8 s[4];
#pragma unroll
    for (int j = 0; j < 4; ++j) {
#pragma unroll
      for (int r = 0; r < 8; ++r) s[j][r] = 0.f;
      const bf16_t* bp = khead + (size_t)(n0 + j * 16 + l15) * CL_ + half * 16;
      bf16x16 bk = ld_frag16(bp, bp + 8);
      s[j] = __builtin_amdgcn_wmma_f32_16x16x32_bf16(false, aq, false, bk,
                                                     (short)0, s[j], false, false);
    }

    // Online softmax (per-lane stats match C-layout rows); P in place of S.
    // Row reductions are single-instruction DPP16 rotations (pure VALU).
#pragma unroll
    for (int r = 0; r < 8; ++r) {
      float rm = fmaxf(fmaxf(s[0][r], s[1][r]), fmaxf(s[2][r], s[3][r]));
      rm = rowred_max(rm);
      float mn = fmaxf(mo[r], rm);
      float esc = __expf(mo[r] - mn);
      mo[r] = mn;
      float rs = 0.f;
#pragma unroll
      for (int j = 0; j < 4; ++j) {
        s[j][r] = __expf(s[j][r] - mn);
        rs += s[j][r];
      }
      rs = rowred_sum(rs);
      lo[r] = lo[r] * esc + rs;
      o0[r] *= esc; o1[r] *= esc;
    }

    // C-layout P -> A-layout via wave-private LDS scratch (in-order per wave)
#pragma unroll
    for (int j = 0; j < 4; ++j)
#pragma unroll
      for (int r = 0; r < 8; ++r)
        plds[(r + half * 8) * 80 + j * 16 + l15] = (bf16_t)s[j][r];

    // O += P*V: 2 k-steps x 2 d-halves = 4 WMMAs; V B-frags direct global
#pragma unroll
    for (int ks = 0; ks < 2; ++ks) {
      const bf16_t* app = plds + l15 * 80 + ks * 32 + half * 16;
      bf16x16 ap = ld_frag16(app, app + 8);
      {
        const bf16_t* vp = vhead + (size_t)(l15) * N_ + n0 + ks * 32 + half * 16;
        bf16x16 bv = ld_frag16(vp, vp + 8);
        o0 = __builtin_amdgcn_wmma_f32_16x16x32_bf16(false, ap, false, bv,
                                                     (short)0, o0, false, false);
      }
      {
        const bf16_t* vp = vhead + (size_t)(16 + l15) * N_ + n0 + ks * 32 + half * 16;
        bf16x16 bv = ld_frag16(vp, vp + 8);
        o1 = __builtin_amdgcn_wmma_f32_16x16x32_bf16(false, ap, false, bv,
                                                     (short)0, o1, false, false);
      }
    }
  }

  // Finalize and store channels-last: ot[b][nq][h*32 + d]
#pragma unroll
  for (int r = 0; r < 8; ++r) {
    float inv = 1.f / lo[r];
    o0[r] *= inv; o1[r] *= inv;
  }
  const size_t obase = (size_t)b * N_ * CL_;
#pragma unroll
  for (int r = 0; r < 8; ++r) {
    int nq = nq0 + wid * 16 + r + half * 8;
    ot[obase + (size_t)nq * CL_ + h * 32 + l15]      = (bf16_t)o0[r];
    ot[obase + (size_t)nq * CL_ + h * 32 + 16 + l15] = (bf16_t)o1[r];
  }
}

// ---------------- host-side orchestration ----------------
extern "C" void kernel_launch(void* const* d_in, const int* in_sizes, int n_in,
                              void* d_out, int out_size, void* d_ws, size_t ws_size,
                              hipStream_t stream) {
  (void)in_sizes; (void)n_in; (void)out_size; (void)ws_size;
  const float* high = (const float*)d_in[0];
  const float* low  = (const float*)d_in[1];
  const float* nh_w = (const float*)d_in[2];
  const float* nh_b = (const float*)d_in[3];
  const float* nl_w = (const float*)d_in[4];
  const float* nl_b = (const float*)d_in[5];
  const float* nf_w = (const float*)d_in[6];
  const float* nf_b = (const float*)d_in[7];
  const float* Wq = (const float*)d_in[8];
  const float* Wk = (const float*)d_in[9];
  const float* Wv = (const float*)d_in[10];
  const float* Wp = (const float*)d_in[11];
  const float* W1 = (const float*)d_in[12];
  const float* W2 = (const float*)d_in[13];
  const float* g_attn = (const float*)d_in[14];
  const float* g_ffn  = (const float*)d_in[15];
  float* outf = (float*)d_out;

  char* base = (char*)d_ws;
  size_t off = 0;
  auto alloc = [&](size_t bytes) -> char* {
    off = (off + 255) & ~(size_t)255;
    char* p = base + off;
    off += bytes;
    return p;
  };
  float*  stats = (float*)alloc(64 * sizeof(float));            // [high:8][low:8][x:8]
  bf16_t* wq    = (bf16_t*)alloc((size_t)CL_ * CH_ * 2);
  bf16_t* wk    = (bf16_t*)alloc((size_t)CL_ * CL_ * 2);
  bf16_t* wv    = (bf16_t*)alloc((size_t)CL_ * CL_ * 2);
  bf16_t* wp    = (bf16_t*)alloc((size_t)CL_ * CL_ * 2);
  bf16_t* w1    = (bf16_t*)alloc((size_t)FFN_ * CL_ * 2);
  bf16_t* w2    = (bf16_t*)alloc((size_t)CL_ * FFN_ * 2);
  bf16_t* highn = (bf16_t*)alloc((size_t)B_ * CH_ * N_ * 2);  // [b][n][256]
  bf16_t* lown  = (bf16_t*)alloc((size_t)B_ * CL_ * N_ * 2);  // [b][n][128]
  bf16_t* qb    = (bf16_t*)alloc((size_t)B_ * CL_ * N_ * 2);  // [b][n][128] (pre-scaled)
  bf16_t* kb    = (bf16_t*)alloc((size_t)B_ * CL_ * N_ * 2);  // [b][n][128]
  bf16_t* vb    = (bf16_t*)alloc((size_t)B_ * CL_ * N_ * 2);  // [b][c][n]
  bf16_t* ob    = (bf16_t*)alloc((size_t)B_ * CL_ * N_ * 2);  // [b][n][128]
  bf16_t* xn    = (bf16_t*)alloc((size_t)B_ * CL_ * N_ * 2);  // [b][n][128]
  bf16_t* h1    = (bf16_t*)alloc((size_t)B_ * FFN_ * N_ * 2); // [b][n][256]

  // Weights -> bf16 (row-major kept: A operand layout)
  cvt_bf16_kernel<<<(CL_ * CH_ + 255) / 256, 256, 0, stream>>>(Wq, wq, CL_ * CH_);
  cvt_bf16_kernel<<<(CL_ * CL_ + 255) / 256, 256, 0, stream>>>(Wk, wk, CL_ * CL_);
  cvt_bf16_kernel<<<(CL_ * CL_ + 255) / 256, 256, 0, stream>>>(Wv, wv, CL_ * CL_);
  cvt_bf16_kernel<<<(CL_ * CL_ + 255) / 256, 256, 0, stream>>>(Wp, wp, CL_ * CL_);
  cvt_bf16_kernel<<<(FFN_ * CL_ + 255) / 256, 256, 0, stream>>>(W1, w1, FFN_ * CL_);
  cvt_bf16_kernel<<<(CL_ * FFN_ + 255) / 256, 256, 0, stream>>>(W2, w2, CL_ * FFN_);

  // GroupNorm stats + normalize to channels-last bf16
  stats_kernel<<<B_, 256, 0, stream>>>(high, CH_ * N_, stats + 0);
  stats_kernel<<<B_, 256, 0, stream>>>(low,  CL_ * N_, stats + 8);
  {
    int tot = B_ * CH_ * N_;
    norm_t_kernel<<<(tot + 255) / 256, 256, 0, stream>>>(high, stats + 0, nh_w, nh_b, highn, CH_, tot);
  }
  {
    int tot = B_ * CL_ * N_;
    norm_t_kernel<<<(tot + 255) / 256, 256, 0, stream>>>(low, stats + 8, nl_w, nl_b, lown, CL_, tot);
  }

  const float qscale = 0.17677669529663687f;  // 1/sqrt(HEAD_DIM)
  // Q (pre-scaled, channels-last), K (channels-last), V (d-major)
  gemm_tn_kernel<1><<<dim3(N_ / 128, CL_ / 64, B_), 256, 0, stream>>>(wq, highn, CL_, CH_, N_, qb, nullptr, nullptr, nullptr, qscale);
  gemm_tn_kernel<1><<<dim3(N_ / 128, CL_ / 64, B_), 256, 0, stream>>>(wk, lown,  CL_, CL_, N_, kb, nullptr, nullptr, nullptr, 1.0f);
  gemm_tn_kernel<0><<<dim3(N_ / 128, CL_ / 64, B_), 256, 0, stream>>>(wv, lown,  CL_, CL_, N_, vb, nullptr, nullptr, nullptr, 1.0f);

  // Flash attention -> ob (channels-last)
  attn_kernel<<<dim3(N_ / 128, B_ * 4), 256, 0, stream>>>(qb, kb, vb, ob);

  // x = low + gamma_attn * (Wproj @ O)  -> d_out (fp32, [b][c][n])
  gemm_tn_kernel<3><<<dim3(N_ / 128, CL_ / 64, B_), 256, 0, stream>>>(wp, ob, CL_, CL_, N_, nullptr, outf, low, g_attn, 1.0f);

  // FFN: norm(x) -> GELU(W1 @ xn) -> out = x + gamma_ffn * (W2 @ h1)
  stats_kernel<<<B_, 256, 0, stream>>>(outf, CL_ * N_, stats + 16);
  {
    int tot = B_ * CL_ * N_;
    norm_t_kernel<<<(tot + 255) / 256, 256, 0, stream>>>(outf, stats + 16, nf_w, nf_b, xn, CL_, tot);
  }
  gemm_tn_kernel<2><<<dim3(N_ / 128, FFN_ / 64, B_), 256, 0, stream>>>(w1, xn, FFN_, CL_, N_, h1, nullptr, nullptr, nullptr, 1.0f);
  gemm_tn_kernel<3><<<dim3(N_ / 128, CL_ / 64, B_), 256, 0, stream>>>(w2, h1, CL_, FFN_, N_, nullptr, outf, outf, g_ffn, 1.0f);
}
